// Decoder_82721070121579
// MI455X (gfx1250) — compile-verified, matrix operand order checked by reading, not run on hardware
//
#include <hip/hip_runtime.h>

typedef __attribute__((ext_vector_type(16))) __bf16 v16bf;
typedef __attribute__((ext_vector_type(8)))  float  v8f;

#define B_  64
#define T_  1000
#define D_  512
#define V_  5000
#define L_  150
#define H_  512
#define E_  512
#define KX  1536      // E + D + H  (concatenated GEMM K for the LSTM gates)
#define NG  2048      // 4*H
#define VP  5008      // V padded to a multiple of 16

__device__ __forceinline__ unsigned short f2bf(float f) {
  unsigned int u = __float_as_uint(f);
  unsigned int r = (u + 0x7FFFu + ((u >> 16) & 1u)) >> 16;
  return (unsigned short)r;
}
__device__ __forceinline__ float sigmf(float x) { return 1.0f / (1.0f + __expf(-x)); }

// ---------------- setup kernels (run once per launch) ----------------

// Wcat[n][k] = bf16( k<1024 ? W_ih[n][k] : W_hh[n][k-1024] ), bcat = b_ih + b_hh
__global__ void pack_wcat(const float* __restrict__ Wih, const float* __restrict__ Whh,
                          const float* __restrict__ bih, const float* __restrict__ bhh,
                          unsigned short* __restrict__ Wcat, float* __restrict__ bcat) {
  for (size_t i = (size_t)blockIdx.x * blockDim.x + threadIdx.x; i < (size_t)NG * KX;
       i += (size_t)gridDim.x * blockDim.x) {
    int n = (int)(i / KX), k = (int)(i % KX);
    float v = (k < 1024) ? Wih[(size_t)n * 1024 + k] : Whh[(size_t)n * 512 + (k - 1024)];
    Wcat[i] = f2bf(v);
    if (i < NG) bcat[i] = bih[i] + bhh[i];
  }
}

__global__ void pack_wout(const float* __restrict__ Wout, unsigned short* __restrict__ Wo) {
  for (size_t i = (size_t)blockIdx.x * blockDim.x + threadIdx.x; i < (size_t)VP * H_;
       i += (size_t)gridDim.x * blockDim.x) {
    int n = (int)(i / H_), k = (int)(i % H_);
    float v = (n < V_) ? Wout[(size_t)n * H_ + k] : 0.0f;
    Wo[i] = f2bf(v);
  }
}

// emb[b][l][e] = bf16( dec_in[b][l] != 0 ? emb_table[dec_in[b][l]][e] : 0 )
__global__ void pack_emb(const float* __restrict__ tab, const int* __restrict__ din,
                         unsigned short* __restrict__ emb) {
  for (size_t i = (size_t)blockIdx.x * blockDim.x + threadIdx.x; i < (size_t)B_ * L_ * E_;
       i += (size_t)gridDim.x * blockDim.x) {
    int e = (int)(i % E_);
    int bl = (int)(i / E_);
    int tok = din[bl];
    float v = (tok != 0) ? tab[(size_t)tok * E_ + e] : 0.0f;
    emb[i] = f2bf(v);
  }
}

__global__ void init_state(float* __restrict__ hf, unsigned short* __restrict__ hbf,
                           float* __restrict__ c) {
  for (size_t i = (size_t)blockIdx.x * blockDim.x + threadIdx.x; i < (size_t)B_ * H_;
       i += (size_t)gridDim.x * blockDim.x) {
    hf[i] = 0.0f; c[i] = 0.0f; hbf[i] = 0;
  }
}

// ---------------- per-step kernels ----------------

// One workgroup per batch row: masked scores -> softmax -> context,
// writes attn to d_out and packs x = [emb_t | ctx | h] in bf16.
__global__ void attn_kernel(const float* __restrict__ enc, const int* __restrict__ enc_len,
                            const float* __restrict__ hf, const unsigned short* __restrict__ hbf,
                            const unsigned short* __restrict__ emb, unsigned short* __restrict__ x,
                            float* __restrict__ attn_out, int t) {
  __shared__ float s_sc[T_];
  __shared__ float s_red[256];
  int b = blockIdx.x, tid = threadIdx.x;
  int len = enc_len[b];
  const float4* hr = (const float4*)(hf + (size_t)b * H_);
  float lmax = -3.4e38f;
  for (int tp = tid; tp < T_; tp += 256) {
    float acc = 0.0f;
    const float4* er = (const float4*)(enc + ((size_t)b * T_ + tp) * D_);
#pragma unroll 4
    for (int q = 0; q < D_ / 4; ++q) {
      float4 e = er[q], h4 = hr[q];
      acc += e.x * h4.x + e.y * h4.y + e.z * h4.z + e.w * h4.w;
    }
    if (tp >= len) acc = -1e9f;
    s_sc[tp] = acc;
    lmax = fmaxf(lmax, acc);
  }
  s_red[tid] = lmax; __syncthreads();
  for (int s = 128; s > 0; s >>= 1) {
    if (tid < s) s_red[tid] = fmaxf(s_red[tid], s_red[tid + s]);
    __syncthreads();
  }
  float mx = s_red[0]; __syncthreads();
  float lsum = 0.0f;
  for (int tp = tid; tp < T_; tp += 256) {
    float e = __expf(s_sc[tp] - mx);
    s_sc[tp] = e; lsum += e;
  }
  s_red[tid] = lsum; __syncthreads();
  for (int s = 128; s > 0; s >>= 1) {
    if (tid < s) s_red[tid] += s_red[tid + s];
    __syncthreads();
  }
  float inv = 1.0f / s_red[0]; __syncthreads();
  for (int tp = tid; tp < T_; tp += 256) {
    float a = s_sc[tp] * inv;
    s_sc[tp] = a;
    attn_out[((size_t)b * L_ + t) * T_ + tp] = a;
  }
  __syncthreads();
  // context: thread owns columns d = tid and tid+256
  float a0 = 0.0f, a1 = 0.0f;
  for (int tp = 0; tp < T_; ++tp) {
    float w = s_sc[tp];
    const float* er = enc + ((size_t)b * T_ + tp) * D_;
    a0 += w * er[tid];
    a1 += w * er[tid + 256];
  }
  unsigned short* xr = x + (size_t)b * KX;
  xr[E_ + tid] = f2bf(a0);
  xr[E_ + tid + 256] = f2bf(a1);
  const unsigned short* e0 = emb + ((size_t)b * L_ + t) * E_;
  xr[tid] = e0[tid];
  xr[tid + 256] = e0[tid + 256];
  xr[1024 + tid] = hbf[(size_t)b * H_ + tid];
  xr[1024 + tid + 256] = hbf[(size_t)b * H_ + tid + 256];
}

// Fused LSTM cell: grid (H/16, B/16), 4 waves/workgroup, wave g computes gate g's
// 16x16 tile via bf16 WMMA over K=1536. The shared A tile (16 rows of x, 48KB
// contiguous) is staged once into LDS with async global->LDS copies; B streams
// from L2. Epilogue does the elementwise cell update through LDS.
__global__ void lstm_gates(const unsigned short* __restrict__ x,
                           const unsigned short* __restrict__ Wcat,
                           const float* __restrict__ bcat, float* __restrict__ c,
                           float* __restrict__ hf, unsigned short* __restrict__ hbf) {
  __shared__ unsigned short s_a[16 * KX];   // 48 KB: A tile in bf16
  __shared__ float g4[4 * 256];
  int ntile = blockIdx.x;        // 0..31 (H/16)
  int mtile = blockIdx.y;        // 0..3  (B/16)
  int tid = threadIdx.x;         // 0..127

  // --- async copy of the contiguous 48KB A block into LDS ---
  {
    const unsigned short* gsrc = x + (size_t)mtile * 16 * KX;
    unsigned lbase = (unsigned)(unsigned long long)(&s_a[0]);
    // 16*KX shorts = 49152 B = 3072 x 16B transfers; 24 per thread (ASYNCcnt <= 24)
    for (int i = tid; i < (16 * KX) / 8; i += 128) {
      unsigned long long ga = (unsigned long long)(gsrc + i * 8);
      unsigned la = lbase + (unsigned)i * 16u;
      asm volatile("global_load_async_to_lds_b128 %0, %1, off"
                   :: "v"(la), "v"(ga) : "memory");
    }
    asm volatile("s_wait_asynccnt 0x0" ::: "memory");
  }
  __syncthreads();

  int w = tid >> 5;              // gate index 0..3 (i,f,g,o)
  int lane = tid & 31;
  int khalf = (lane >> 4) * 16;
  int nrow = w * 512 + ntile * 16 + (lane & 15);
  const unsigned short* arow = &s_a[(lane & 15) * KX + khalf];   // LDS reads
  const unsigned short* brow = Wcat + (size_t)nrow * KX + khalf;
  v8f acc = {};
  for (int k0 = 0; k0 < KX; k0 += 32) {
    v16bf a = *(const v16bf*)(arow + k0);
    v16bf b = *(const v16bf*)(brow + k0);
    __builtin_prefetch(brow + k0 + 128, 0, 0);
    acc = __builtin_amdgcn_wmma_f32_16x16x32_bf16(false, a, false, b, (short)0, acc,
                                                  false, false);
  }
  int mbase = (lane >> 4) * 8;
#pragma unroll
  for (int r = 0; r < 8; ++r)
    g4[w * 256 + (mbase + r) * 16 + (lane & 15)] = acc[r];
  __syncthreads();
  for (int e = tid; e < 256; e += 128) {
    int ml = e >> 4, nl = e & 15;
    int bidx = mtile * 16 + ml;
    int hcol = ntile * 16 + nl;
    float gi = g4[0 * 256 + e] + bcat[0 * 512 + hcol];
    float gf = g4[1 * 256 + e] + bcat[1 * 512 + hcol];
    float gg = g4[2 * 256 + e] + bcat[2 * 512 + hcol];
    float go = g4[3 * 256 + e] + bcat[3 * 512 + hcol];
    size_t idx = (size_t)bidx * H_ + hcol;
    float c2 = sigmf(gf) * c[idx] + sigmf(gi) * tanhf(gg);
    float h2 = sigmf(go) * tanhf(c2);
    c[idx] = c2;
    hf[idx] = h2;
    hbf[idx] = f2bf(h2);
  }
}

// Output projection: grid (VP/16, B/16), one wave per 16x16 tile, K=512 bf16 WMMA.
__global__ void logits_kernel(const unsigned short* __restrict__ hbf,
                              const unsigned short* __restrict__ Wo,
                              const float* __restrict__ bout, float* __restrict__ out, int t) {
  int ntile = blockIdx.x;        // 0..312
  int mtile = blockIdx.y;        // 0..3
  int lane = threadIdx.x;        // 0..31
  int mrow = mtile * 16 + (lane & 15);
  int khalf = (lane >> 4) * 16;
  int nrow = ntile * 16 + (lane & 15);
  const unsigned short* arow = hbf + (size_t)mrow * H_ + khalf;
  const unsigned short* brow = Wo + (size_t)nrow * H_ + khalf;
  v8f acc = {};
  for (int k0 = 0; k0 < H_; k0 += 32) {
    v16bf a = *(const v16bf*)(arow + k0);
    v16bf b = *(const v16bf*)(brow + k0);
    acc = __builtin_amdgcn_wmma_f32_16x16x32_bf16(false, a, false, b, (short)0, acc,
                                                  false, false);
  }
  int mbase = (lane >> 4) * 8;
  int nl = lane & 15;
  int v = ntile * 16 + nl;
  if (v < V_) {
    float bo = bout[v];
#pragma unroll
    for (int r = 0; r < 8; ++r) {
      int bidx = mtile * 16 + mbase + r;
      out[((size_t)bidx * L_ + t) * V_ + v] = acc[r] + bo;
    }
  }
}

// log-softmax scalar (at target), argmax — one workgroup per batch row.
__global__ void lsm_kernel(const float* __restrict__ logits, const int* __restrict__ dout,
                           float* __restrict__ dlp, float* __restrict__ pred, int t) {
  __shared__ float s_v[256];
  __shared__ int s_i[256];
  int b = blockIdx.x, tid = threadIdx.x;
  const float* row = logits + ((size_t)b * L_ + t) * V_;
  float bm = -3.4e38f; int bi = 0;
  for (int v = tid; v < V_; v += 256) {
    float xv = row[v];
    if (xv > bm) { bm = xv; bi = v; }
  }
  s_v[tid] = bm; s_i[tid] = bi; __syncthreads();
  for (int s = 128; s > 0; s >>= 1) {
    if (tid < s) {
      float ov = s_v[tid + s]; int oi = s_i[tid + s];
      if (ov > s_v[tid] || (ov == s_v[tid] && oi < s_i[tid])) { s_v[tid] = ov; s_i[tid] = oi; }
    }
    __syncthreads();
  }
  float mx = s_v[0]; int am = s_i[0]; __syncthreads();
  float ls = 0.0f;
  for (int v = tid; v < V_; v += 256) ls += __expf(row[v] - mx);
  s_v[tid] = ls; __syncthreads();
  for (int s = 128; s > 0; s >>= 1) {
    if (tid < s) s_v[tid] += s_v[tid + s];
    __syncthreads();
  }
  if (tid == 0) {
    float lse = mx + __logf(s_v[0]);
    int tgt = dout[b * L_ + t];
    dlp[b * L_ + t] = row[tgt] - lse;
    pred[b * L_ + t] = (float)am;
  }
}

extern "C" void kernel_launch(void* const* d_in, const int* in_sizes, int n_in,
                              void* d_out, int out_size, void* d_ws, size_t ws_size,
                              hipStream_t stream) {
  (void)in_sizes; (void)n_in; (void)out_size; (void)ws_size;
  const float* enc  = (const float*)d_in[0];
  const int*   elen = (const int*)d_in[1];
  const int*   din  = (const int*)d_in[2];
  const int*   dou  = (const int*)d_in[3];
  const float* tab  = (const float*)d_in[4];
  const float* Wih  = (const float*)d_in[5];
  const float* bih  = (const float*)d_in[6];
  const float* Whh  = (const float*)d_in[7];
  const float* bhh  = (const float*)d_in[8];
  const float* Wout = (const float*)d_in[9];
  const float* bout = (const float*)d_in[10];

  char* ws = (char*)d_ws;
  size_t off = 0;
  auto alloc = [&](size_t bytes) -> void* {
    void* p = ws + off;
    off = (off + bytes + 255) & ~(size_t)255;
    return p;
  };
  unsigned short* Wcat = (unsigned short*)alloc((size_t)NG * KX * 2);
  unsigned short* Wo   = (unsigned short*)alloc((size_t)VP * H_ * 2);
  unsigned short* emb  = (unsigned short*)alloc((size_t)B_ * L_ * E_ * 2);
  unsigned short* x    = (unsigned short*)alloc((size_t)B_ * KX * 2);
  unsigned short* hbf  = (unsigned short*)alloc((size_t)B_ * H_ * 2);
  float* hf   = (float*)alloc((size_t)B_ * H_ * 4);
  float* c    = (float*)alloc((size_t)B_ * H_ * 4);
  float* bcat = (float*)alloc((size_t)NG * 4);

  float* out_logits = (float*)d_out;
  float* out_dlp    = out_logits + (size_t)B_ * L_ * V_;
  float* out_pred   = out_dlp + (size_t)B_ * L_;
  float* out_attn   = out_pred + (size_t)B_ * L_;

  pack_wcat<<<2048, 256, 0, stream>>>(Wih, Whh, bih, bhh, Wcat, bcat);
  pack_wout<<<2048, 256, 0, stream>>>(Wout, Wo);
  pack_emb<<<2048, 256, 0, stream>>>(tab, din, emb);
  init_state<<<128, 256, 0, stream>>>(hf, hbf, c);

  for (int t = 0; t < L_; ++t) {
    attn_kernel<<<B_, 256, 0, stream>>>(enc, elen, hf, hbf, emb, x, out_attn, t);
    lstm_gates<<<dim3(H_ / 16, B_ / 16), 128, 0, stream>>>(x, Wcat, bcat, c, hf, hbf);
    logits_kernel<<<dim3(VP / 16, B_ / 16), 32, 0, stream>>>(hbf, Wo, bout, out_logits, t);
    lsm_kernel<<<B_, 256, 0, stream>>>(out_logits, dou, out_dlp, out_pred, t);
  }
}
